// ConvTranspose2D_7559142441254
// MI455X (gfx1250) — compile-verified, matrix operand order checked by reading, not run on hardware
//
#include <hip/hip_runtime.h>
#include <hip/hip_bf16.h>

typedef __attribute__((ext_vector_type(16))) _Float16 v16h;
typedef __attribute__((ext_vector_type(8)))  float    v8f;

#define CIN   256
#define COUT  128
#define HIN   128
#define HOUT  256
#define XPAD_W 130                      // 128 + halo 1 each side
#define XPAD_CH (XPAD_W * XPAD_W)       // per-channel padded plane (halves)
#define XPAD_HALVES (CIN * XPAD_CH)     // 4,326,400
#define WPACK_HALVES (4 * 8 * 32 * 32 * 16)  // 524,288 (parity,mtile,kstep,lane,idx)

// ---- pack x -> f16 with zero halo -------------------------------------------
__global__ void pack_x_f16(const float* __restrict__ x, _Float16* __restrict__ xpad) {
    int i = blockIdx.x * blockDim.x + threadIdx.x;
    if (i >= XPAD_HALVES) return;
    int ch = i / XPAD_CH;
    int rem = i - ch * XPAD_CH;
    int r = rem / XPAD_W;
    int c = rem - r * XPAD_W;
    float v = 0.f;
    if (r >= 1 && r <= HIN && c >= 1 && c <= HIN)
        v = x[(ch * HIN + (r - 1)) * HIN + (c - 1)];
    xpad[i] = (_Float16)v;
}

// ---- pack W -> f16 in WMMA A-fragment layout --------------------------------
// K index = ci*4 + ty*2 + tx ; kh = p + 2*ty ; kw = q + 2*tx
// A 16x32 f16 layout: lane<16 : M=lane,   idx<8 -> K=idx,   idx>=8 -> K=idx+8
//                     lane>=16: M=lane-16,idx<8 -> K=idx+8, idx>=8 -> K=idx+16
__global__ void pack_w_f16(const float* __restrict__ W, _Float16* __restrict__ wpack) {
    int f = blockIdx.x * blockDim.x + threadIdx.x;
    if (f >= WPACK_HALVES) return;
    int idx    = f & 15;
    int lane   = (f >> 4)  & 31;
    int kstep  = (f >> 9)  & 31;
    int mtile  = (f >> 14) & 7;
    int parity = f >> 17;                 // 0..3 = (p<<1)|q
    int p = parity >> 1, q = parity & 1;
    int grp = lane >> 4;
    int klocal = (idx & 7) + ((idx >> 3) << 4) + (grp << 3);
    int K  = kstep * 32 + klocal;         // 0..1023
    int ci = K >> 2;
    int ty = (K >> 1) & 1;
    int tx = K & 1;
    int kh = p + 2 * ty;
    int kw = q + 2 * tx;
    int co = mtile * 16 + (lane & 15);
    float v = W[((co * CIN + ci) << 4) + kh * 4 + kw];
    wpack[f] = (_Float16)v;
}

// ---- main implicit-GEMM kernel ----------------------------------------------
// block = 256 thr = 8 waves; wave w -> M-tile w (co 16w..16w+15)
// block covers M=128 x N=32 (one spatial row y, cols x0..x0+31), one parity.
// Each wave: 2 N-subtiles sharing one A fragment per k-step.
// blockIdx.x = parity*512 + ntile ; ntile -> y = nt>>2, x0 = (nt&3)*32
__global__ __launch_bounds__(256)
void convT_wmma(const _Float16* __restrict__ xpad,
                const _Float16* __restrict__ wpack,
                const float* __restrict__ bias,
                float* __restrict__ out) {
    // 64 KB: [nsub(2)][kstep(32)][lane(32)][idx(16)] halves
    __shared__ __align__(32) _Float16 Bsh[2 * 32 * 512];

    int blk    = blockIdx.x;
    int parity = blk >> 9;
    int ntile  = blk & 511;
    int p = parity >> 1, q = parity & 1;
    int y  = ntile >> 2;
    int x0 = (ntile & 3) << 5;
    int tid = threadIdx.x;

    // Build B panel (K=1024 x N=32) in fragment layout: 128 values/thread.
    // value(kg, n) = xpad[ci][y + p + ty][x0 + n + q + tx]
    const _Float16* xrow = xpad + (y + p) * XPAD_W + (x0 + q);
#pragma unroll 8
    for (int j = 0; j < 128; ++j) {
        int vi = j * 256 + tid;           // = kg*32 + n
        int n  = vi & 31;
        int kg = vi >> 5;                 // 0..1023
        int ci = kg >> 2;
        int ty = (kg >> 1) & 1;
        int tx = kg & 1;
        _Float16 v = xrow[ci * XPAD_CH + ty * XPAD_W + n + tx];
        int ks = kg >> 5;
        int k  = kg & 31;
        int nsub  = n >> 4;
        int flane = (n & 15) + (((k >> 3) & 1) << 4);
        int fidx  = (k & 7) + ((k >> 4) << 3);
        Bsh[((nsub * 32 + ks) * 32 + flane) * 16 + fidx] = v;
    }
    __syncthreads();

    int wave = tid >> 5;
    int lane = tid & 31;
    const _Float16* Aptr = wpack + ((parity * 8 + wave) * 16384) + lane * 16;

    v8f c0 = {};
    v8f c1 = {};
#pragma unroll 4
    for (int ks = 0; ks < 32; ++ks) {
        v16h a  = *(const v16h*)(Aptr + ks * 512);
        v16h b0 = *(const v16h*)(&Bsh[(ks * 32 + lane) * 16]);
        v16h b1 = *(const v16h*)(&Bsh[((32 + ks) * 32 + lane) * 16]);
        c0 = __builtin_amdgcn_wmma_f32_16x16x32_f16(
                 false, a, false, b0, (short)0, c0, false, false);
        c1 = __builtin_amdgcn_wmma_f32_16x16x32_f16(
                 false, a, false, b1, (short)0, c1, false, false);
    }

    // store: lane<16 -> co rows 0..7, lane>=16 -> co rows 8..15 ; N = lane&15
    int co_base = wave * 16 + ((lane >> 4) << 3);
    int nl = lane & 15;
    int oh  = (y << 1) + p;
    int ow0 = ((x0 + nl) << 1) + q;
    int ow1 = ((x0 + 16 + nl) << 1) + q;
    float* ob0 = out + (size_t)oh * HOUT + ow0;
    float* ob1 = out + (size_t)oh * HOUT + ow1;
#pragma unroll
    for (int v = 0; v < 8; ++v) {
        int co = co_base + v;
        float bv = bias[co];
        ob0[(size_t)co * (HOUT * HOUT)] = c0[v] + bv;
        ob1[(size_t)co * (HOUT * HOUT)] = c1[v] + bv;
    }
}

extern "C" void kernel_launch(void* const* d_in, const int* in_sizes, int n_in,
                              void* d_out, int out_size, void* d_ws, size_t ws_size,
                              hipStream_t stream) {
    const float* x = (const float*)d_in[0];
    const float* W = (const float*)d_in[1];
    const float* b = (const float*)d_in[2];
    float* out = (float*)d_out;

    _Float16* xpad  = (_Float16*)d_ws;
    _Float16* wpack = (_Float16*)((char*)d_ws + (size_t)XPAD_HALVES * 2);

    pack_x_f16<<<(XPAD_HALVES + 255) / 256, 256, 0, stream>>>(x, xpad);
    pack_w_f16<<<(WPACK_HALVES + 255) / 256, 256, 0, stream>>>(W, wpack);
    convT_wmma<<<2048, 256, 0, stream>>>(xpad, wpack, b, out);
}
